// GPT2MoLExFORWARD_79018808312207
// MI455X (gfx1250) — compile-verified, matrix operand order checked by reading, not run on hardware
//
#include <hip/hip_runtime.h>
#include <math.h>
#include <cstdint>

typedef __bf16 bf16_t;
typedef __attribute__((ext_vector_type(16))) __bf16 v16bf;
typedef __attribute__((ext_vector_type(8)))  __bf16 v8bf;
typedef __attribute__((ext_vector_type(8)))  float  v8f;

#define TOK     1024      // B*T
#define DMODEL  768
#define TSEQ    512
#define NH      12
#define HD      64
#define PROJ_   256
#define NL      6
#define CLAMP_MAX_F 4.60517018598809136804f   // ln(100)

__device__ __forceinline__ float gelu_f(float x) {
  return 0.5f * x * (1.0f + tanhf(0.7978845608028654f * (x + 0.044715f * x * x * x)));
}

// CDNA5 async global->LDS copy (ASYNCcnt-tracked, bypasses VGPRs).
// INST_OFFSET is added to BOTH the LDS and global address (ISA async pseudocode).
__device__ __forceinline__ void async_ld_b128(unsigned lds, const void* g) {
  asm volatile("global_load_async_to_lds_b128 %0, %1, off" :: "v"(lds), "v"(g) : "memory");
}
__device__ __forceinline__ void async_ld_b128_o16(unsigned lds, const void* g) {
  asm volatile("global_load_async_to_lds_b128 %0, %1, off offset:16" :: "v"(lds), "v"(g) : "memory");
}

// ---------------------------------------------------------------------------
// Batched WMMA GEMM over experts (gridDim.z):
//   out_e[M,N] = act(A_e[M,K](bf16) @ Wt_e[N,K](bf16)^T + bias_e + residual_e)
// Wt pre-transposed to [N,K] so both tiles stage contiguously.
// 256 threads = 8 waves (4x2), block tile 128x64, wave tile 32x32 (2x2 WMMA).
// Double-buffered LDS fed by global_load_async_to_lds_b128.
// ---------------------------------------------------------------------------
#define BM 128
#define BN 64
#define BK 32
#define LSTR 40   // padded LDS stride in bf16 elements (80B rows, 16B aligned)

__global__ __launch_bounds__(256) void wmma_gemm_k(
    const bf16_t* __restrict__ A, long As_e,
    const bf16_t* __restrict__ Wt, long Wt_e,
    const float* __restrict__ bias, long bias_e,
    const float* __restrict__ residual, long res_e,
    float* __restrict__ outF, bf16_t* __restrict__ outB, long out_e,
    int K, int N, int act)
{
  __shared__ bf16_t As[2][BM * LSTR];
  __shared__ bf16_t Bs[2][BN * LSTR];   // n-major
  const int e = blockIdx.z;
  A  += (size_t)e * As_e;
  Wt += (size_t)e * Wt_e;
  if (bias)     bias     += (size_t)e * bias_e;
  if (residual) residual += (size_t)e * res_e;
  if (outF)     outF     += (size_t)e * out_e;
  if (outB)     outB     += (size_t)e * out_e;

  const int tid  = threadIdx.x;
  const int wave = tid >> 5;
  const int lane = tid & 31;
  const int l16  = lane & 15;
  const int lhi  = lane >> 4;
  const int wm   = wave & 3;         // row slice 0..3
  const int wn   = wave >> 2;        // col slice 0..1
  const int row0 = blockIdx.y * BM;
  const int col0 = blockIdx.x * BN;

  v8f acc[2][2] = {};

  // staging map: A: 2 x b128 per lane; B: 1 x b128 per lane  => 3 async/tile/wave
  const int ar  = tid >> 1;          // 0..127
  const int ak  = (tid & 1) * 16;    // 0/16
  const int bn  = tid >> 2;          // 0..63
  const int bkc = (tid & 3) * 8;     // 0,8,16,24

  const bf16_t* gA = A  + (size_t)(row0 + ar) * K + ak;
  const bf16_t* gB = Wt + (size_t)(col0 + bn) * K + bkc;
  unsigned ldsA[2], ldsB[2];
  ldsA[0] = (unsigned)reinterpret_cast<uintptr_t>(&As[0][ar * LSTR + ak]);
  ldsA[1] = (unsigned)reinterpret_cast<uintptr_t>(&As[1][ar * LSTR + ak]);
  ldsB[0] = (unsigned)reinterpret_cast<uintptr_t>(&Bs[0][bn * LSTR + bkc]);
  ldsB[1] = (unsigned)reinterpret_cast<uintptr_t>(&Bs[1][bn * LSTR + bkc]);

  // prologue: tile 0 -> buffer 0
  async_ld_b128(ldsA[0], gA);
  async_ld_b128_o16(ldsA[0], gA);
  async_ld_b128(ldsB[0], gB);

  const int nk = K / BK;
  for (int t = 0; t < nk; ++t) {
    const int p = t & 1;
    if (t + 1 < nk) {           // prefetch next tile into the other buffer
      const bf16_t* nA = gA + (size_t)(t + 1) * BK;
      const bf16_t* nB = gB + (size_t)(t + 1) * BK;
      async_ld_b128(ldsA[p ^ 1], nA);
      async_ld_b128_o16(ldsA[p ^ 1], nA);
      async_ld_b128(ldsB[p ^ 1], nB);
      // async loads complete in order: <=3 outstanding => current tile landed
      asm volatile("s_wait_asynccnt 0x3" ::: "memory");
    } else {
      asm volatile("s_wait_asynccnt 0x0" ::: "memory");
    }
    __syncthreads();            // all waves' chunks of tile p are in LDS

    const bf16_t* Ap = &As[p][0];
    const bf16_t* Bp = &Bs[p][0];
    #pragma unroll
    for (int ti = 0; ti < 2; ++ti) {
      const int m = wm * 32 + ti * 16 + l16;
      // ISA 16-bit A 16x32 layout: lane half picks k-block of 8; vgprs 4-7 = k+16
      v8bf alo = *(const v8bf*)&Ap[m * LSTR + lhi * 8];
      v8bf ahi = *(const v8bf*)&Ap[m * LSTR + 16 + lhi * 8];
      v16bf afrag;
      #pragma unroll
      for (int i = 0; i < 8; ++i) { afrag[i] = alo[i]; afrag[i + 8] = ahi[i]; }
      #pragma unroll
      for (int tj = 0; tj < 2; ++tj) {
        const int n = wn * 32 + tj * 16 + l16;
        v8bf blo = *(const v8bf*)&Bp[n * LSTR + lhi * 8];
        v8bf bhi = *(const v8bf*)&Bp[n * LSTR + 16 + lhi * 8];
        v16bf bfrag;
        #pragma unroll
        for (int i = 0; i < 8; ++i) { bfrag[i] = blo[i]; bfrag[i + 8] = bhi[i]; }
        acc[ti][tj] = __builtin_amdgcn_wmma_f32_16x16x32_bf16(
            false, afrag, false, bfrag, (short)0, acc[ti][tj], false, false);
      }
    }
    __syncthreads();            // done reading tile p before it is overwritten
  }

  // C/D layout: lane l, vgpr r -> row = r + 8*(l>>4), col = l&15
  #pragma unroll
  for (int ti = 0; ti < 2; ++ti) {
    #pragma unroll
    for (int tj = 0; tj < 2; ++tj) {
      #pragma unroll
      for (int r = 0; r < 8; ++r) {
        const int row = row0 + wm * 32 + ti * 16 + 8 * lhi + r;
        const int col = col0 + wn * 32 + tj * 16 + l16;
        float v = acc[ti][tj][r];
        if (bias)     v += bias[col];
        if (residual) v += residual[(size_t)row * N + col];
        if (act == 1) v = gelu_f(v);
        const size_t o = (size_t)row * N + col;
        if (outF) outF[o] = v;
        if (outB) outB[o] = (bf16_t)v;
      }
    }
  }
}

// ---------------------------------------------------------------------------
// Causal MHA from packed qkv [E][TOK,2304] fp32, bf16 out [E][TOK,DMODEL].
// grid (TSEQ/32, NH, B*E), 256 threads; each wave owns one q row at a time.
// ---------------------------------------------------------------------------
__global__ __launch_bounds__(256) void attn_k(const float* __restrict__ qkv,
                                              bf16_t* __restrict__ outB)
{
  __shared__ float sc[8][TSEQ];
  __shared__ float qs[8][HD];
  const int e    = blockIdx.z >> 1;
  const int b    = blockIdx.z & 1;
  qkv  += (size_t)e * TOK * 2304;
  outB += (size_t)e * TOK * DMODEL;
  const int hh   = blockIdx.y;
  const int qb   = blockIdx.x;
  const int wave = threadIdx.x >> 5;
  const int lane = threadIdx.x & 31;
  const int kmax = qb * 32 + 32;     // uniform key bound for this q block

  for (int rr = 0; rr < 4; ++rr) {
    const int q = qb * 32 + rr * 8 + wave;
    const size_t qoff = (size_t)(b * TSEQ + q) * 2304 + (size_t)hh * HD;
    qs[wave][lane]      = qkv[qoff + lane];
    qs[wave][lane + 32] = qkv[qoff + lane + 32];
    __syncthreads();

    for (int kk = lane; kk < kmax; kk += 32) {
      const float* kp = qkv + (size_t)(b * TSEQ + kk) * 2304 + 768 + (size_t)hh * HD;
      float s = 0.f;
      #pragma unroll
      for (int d = 0; d < HD; d += 4) {
        float4 kv = *(const float4*)(kp + d);
        s += qs[wave][d] * kv.x + qs[wave][d + 1] * kv.y
           + qs[wave][d + 2] * kv.z + qs[wave][d + 3] * kv.w;
      }
      sc[wave][kk] = (kk <= q) ? s * 0.125f : -1e30f;
    }
    __syncthreads();

    float m = -1e30f;
    for (int kk = lane; kk < kmax; kk += 32) m = fmaxf(m, sc[wave][kk]);
    #pragma unroll
    for (int o = 16; o > 0; o >>= 1) m = fmaxf(m, __shfl_xor(m, o, 32));
    float ssum = 0.f;
    for (int kk = lane; kk < kmax; kk += 32) {
      float p = __expf(sc[wave][kk] - m);
      sc[wave][kk] = p;
      ssum += p;
    }
    #pragma unroll
    for (int o = 16; o > 0; o >>= 1) ssum += __shfl_xor(ssum, o, 32);
    const float inv = 1.0f / ssum;
    __syncthreads();

    float o0 = 0.f, o1 = 0.f;
    for (int kk = 0; kk < kmax; ++kk) {
      const float p = sc[wave][kk];
      const float* vp = qkv + (size_t)(b * TSEQ + kk) * 2304 + 1536 + (size_t)hh * HD;
      o0 += p * vp[lane];
      o1 += p * vp[lane + 32];
    }
    const size_t ooff = (size_t)(b * TSEQ + q) * DMODEL + (size_t)hh * HD;
    outB[ooff + lane]      = (bf16_t)(o0 * inv);
    outB[ooff + lane + 32] = (bf16_t)(o1 * inv);
    __syncthreads();
  }
}

// ---------------------------------------------------------------------------
// Batched LayerNorm. grid (TOK, E). x_e/p_e/out_e = per-expert strides
// (x_e=0 -> shared input; p_e=DMODEL -> per-layer params). bf16/fp32 out.
// ---------------------------------------------------------------------------
__global__ __launch_bounds__(256) void layernorm_k(const float* __restrict__ x, long x_e,
    const float* __restrict__ g, const float* __restrict__ b, long p_e,
    bf16_t* __restrict__ outB, float* __restrict__ outF, long out_e)
{
  __shared__ float red[256];
  const int e = blockIdx.y;
  const int t = blockIdx.x;
  const float* xr = x + (size_t)e * x_e + (size_t)t * DMODEL;
  const float* ge = g + (size_t)e * p_e;
  const float* be = b + (size_t)e * p_e;
  float s = 0.f;
  for (int i = threadIdx.x; i < DMODEL; i += 256) s += xr[i];
  red[threadIdx.x] = s; __syncthreads();
  for (int st = 128; st > 0; st >>= 1) {
    if (threadIdx.x < st) red[threadIdx.x] += red[threadIdx.x + st];
    __syncthreads();
  }
  const float mean = red[0] * (1.0f / DMODEL);
  __syncthreads();
  float v = 0.f;
  for (int i = threadIdx.x; i < DMODEL; i += 256) { float d = xr[i] - mean; v += d * d; }
  red[threadIdx.x] = v; __syncthreads();
  for (int st = 128; st > 0; st >>= 1) {
    if (threadIdx.x < st) red[threadIdx.x] += red[threadIdx.x + st];
    __syncthreads();
  }
  const float inv = rsqrtf(red[0] * (1.0f / DMODEL) + 1e-5f);
  for (int i = threadIdx.x; i < DMODEL; i += 256) {
    float y = ge[i] * (xr[i] - mean) * inv + be[i];
    const size_t o = (size_t)e * out_e + (size_t)t * DMODEL + i;
    if (outB) outB[o] = (bf16_t)y;
    if (outF) outF[o] = y;
  }
}

// Embedding: h[row] = wte[ids[row]] + wpe[row % T]
__global__ __launch_bounds__(256) void embed_k(const int* __restrict__ ids,
    const float* __restrict__ wte, const float* __restrict__ wpe,
    float* __restrict__ h)
{
  const int row = blockIdx.x;
  const int t   = row & (TSEQ - 1);
  const int id  = ids[row];
  for (int d = threadIdx.x; d < DMODEL; d += 256)
    h[(size_t)row * DMODEL + d] = wte[(size_t)id * DMODEL + d] + wpe[(size_t)t * DMODEL + d];
}

// fp32 -> bf16 grid-stride convert
__global__ __launch_bounds__(256) void cvt_bf16_k(const float* __restrict__ in,
                                                  bf16_t* __restrict__ out, long n)
{
  for (long i = (long)blockIdx.x * 256 + threadIdx.x; i < n; i += (long)gridDim.x * 256)
    out[i] = (bf16_t)in[i];
}

// Tiled transpose+convert: W fp32 [mat][K][N] -> Wt bf16 [mat][N][K].
// grid (N/32, K/32, nmat), block 256.
__global__ __launch_bounds__(256) void w_tr_bf16_k(const float* __restrict__ W,
    bf16_t* __restrict__ Wt, int K, int N)
{
  __shared__ float tile[32][33];
  const long mat = blockIdx.z;
  const float* Wm = W  + mat * (long)K * N;
  bf16_t*     Wtm = Wt + mat * (long)K * N;
  const int k0 = blockIdx.y * 32;
  const int n0 = blockIdx.x * 32;
  const int c  = threadIdx.x & 31;
  const int r0 = threadIdx.x >> 5;   // 0..7
  #pragma unroll
  for (int i = 0; i < 4; ++i) {
    const int r = r0 + i * 8;
    tile[r][c] = Wm[(size_t)(k0 + r) * N + (n0 + c)];
  }
  __syncthreads();
  #pragma unroll
  for (int i = 0; i < 4; ++i) {
    const int r = r0 + i * 8;        // local n row
    Wtm[(size_t)(n0 + r) * K + (k0 + c)] = (bf16_t)tile[c][r];
  }
}

// Column-normalize gate_sim[:, :E]. grid 1, block 256 (= PROJ). simn stride 6.
__global__ __launch_bounds__(256) void gate_sim_norm_k(const float* __restrict__ sim,
                                                       float* __restrict__ simn, int E)
{
  __shared__ float red[256];
  const int p = threadIdx.x;
  for (int e = 0; e < E; ++e) {
    const float vv = sim[p * NL + e];
    red[p] = vv * vv; __syncthreads();
    for (int st = 128; st > 0; st >>= 1) {
      if (p < st) red[p] += red[p + st];
      __syncthreads();
    }
    const float scale = 1.0f / fmaxf(sqrtf(red[0]), 1e-12f);
    __syncthreads();
    simn[p * NL + e] = vv * scale;
  }
}

// Per-token gate: normalize proj row, logits = projn @ simn * exp(min(temp,clamp)),
// softmax + argmax. One wave per token, grid TOK/8.
__global__ __launch_bounds__(256) void gate_finalize_k(const float* __restrict__ proj,
    const float* __restrict__ simn, const float* __restrict__ gate_temp, int ind, int E,
    int* __restrict__ idxo, float* __restrict__ gvalo)
{
  const int wave  = threadIdx.x >> 5;
  const int lane  = threadIdx.x & 31;
  const int token = blockIdx.x * 8 + wave;
  const float* pr = proj + (size_t)token * PROJ_;
  float pv[8];
  float ss = 0.f;
  #pragma unroll
  for (int i = 0; i < 8; ++i) { pv[i] = pr[lane + i * 32]; ss += pv[i] * pv[i]; }
  #pragma unroll
  for (int o = 16; o > 0; o >>= 1) ss += __shfl_xor(ss, o, 32);
  const float inv    = 1.0f / fmaxf(sqrtf(ss), 1e-12f);
  const float tscale = __expf(fminf(gate_temp[ind], CLAMP_MAX_F));
  float lg[NL];
  for (int e = 0; e < E; ++e) {
    float d = 0.f;
    #pragma unroll
    for (int i = 0; i < 8; ++i) d += pv[i] * simn[(size_t)(lane + i * 32) * NL + e];
    #pragma unroll
    for (int o = 16; o > 0; o >>= 1) d += __shfl_xor(d, o, 32);
    lg[e] = d * inv * tscale;
  }
  float mx = -1e30f; int am = 0;
  for (int e = 0; e < E; ++e) if (lg[e] > mx) { mx = lg[e]; am = e; }
  float se = 0.f;
  for (int e = 0; e < E; ++e) se += __expf(lg[e] - mx);
  if (lane == 0) { idxo[token] = am; gvalo[token] = 1.0f / se; }  // prob at argmax
}

// h = w_main*outs[0] + w_other*gval*outs[idx]
__global__ __launch_bounds__(256) void combine_k(const float* __restrict__ outs,
    const int* __restrict__ idxv, const float* __restrict__ gval,
    const float* __restrict__ w_main, const float* __restrict__ w_other, int ind,
    float* __restrict__ h)
{
  const float wmv = w_main[ind], wov = w_other[ind];
  const size_t n = (size_t)TOK * DMODEL;
  for (size_t i = (size_t)blockIdx.x * 256 + threadIdx.x; i < n; i += (size_t)gridDim.x * 256) {
    const size_t row = i / DMODEL;
    const float o0 = outs[i];
    const float oe = outs[(size_t)idxv[row] * TOK * DMODEL + i];
    h[i] = wmv * o0 + wov * gval[row] * oe;
  }
}

// ---------------------------------------------------------------------------
extern "C" void kernel_launch(void* const* d_in, const int* in_sizes, int n_in,
                              void* d_out, int out_size, void* d_ws, size_t ws_size,
                              hipStream_t stream) {
  (void)in_sizes; (void)n_in; (void)out_size;
  const int*   ids      = (const int*)  d_in[0];
  const float* wte      = (const float*)d_in[1];
  const float* wpe      = (const float*)d_in[2];
  const float* ln1_g    = (const float*)d_in[3];
  const float* ln1_b    = (const float*)d_in[4];
  const float* attn_w   = (const float*)d_in[5];
  const float* attn_b   = (const float*)d_in[6];
  const float* cproj_w  = (const float*)d_in[7];
  const float* cproj_b  = (const float*)d_in[8];
  const float* ln2_g    = (const float*)d_in[9];
  const float* ln2_b    = (const float*)d_in[10];
  const float* fc_w     = (const float*)d_in[11];
  const float* fc_b     = (const float*)d_in[12];
  const float* fc2_w    = (const float*)d_in[13];
  const float* fc2_b    = (const float*)d_in[14];
  const float* gate_w   = (const float*)d_in[15];
  const float* gate_b   = (const float*)d_in[16];
  const float* gate_sim = (const float*)d_in[17];
  const float* gate_temp= (const float*)d_in[18];
  const float* w_main   = (const float*)d_in[19];
  const float* w_other  = (const float*)d_in[20];
  const float* lnf_g    = (const float*)d_in[21];
  const float* lnf_b    = (const float*)d_in[22];

  char* ws = (char*)d_ws;
  size_t off = 0;
  auto alloc = [&](size_t bytes) -> char* {
    off = (off + 255) & ~(size_t)255;
    char* p = ws + off;
    off += bytes;
    return p;
  };

  // ---- fixed allocations ----
  // bf16 [N,K] weight cache (~87 MB total -> L2-resident after first touch)
  bf16_t* attn_wt_bf  = (bf16_t*)alloc(6UL * DMODEL * 2304 * 2);
  bf16_t* cproj_wt_bf = (bf16_t*)alloc(6UL * DMODEL * DMODEL * 2);
  bf16_t* fc_wt_bf    = (bf16_t*)alloc(6UL * DMODEL * 3072 * 2);
  bf16_t* fc2_wt_bf   = (bf16_t*)alloc(6UL * 3072 * DMODEL * 2);
  bf16_t* gate_wt_bf  = (bf16_t*)alloc(5UL * PROJ_ * DMODEL * 2);  // already [N,K]
  float*  h        = (float*) alloc((size_t)TOK * DMODEL * 4);
  bf16_t* h_bf     = (bf16_t*)alloc((size_t)TOK * DMODEL * 2);
  float*  outs     = (float*) alloc(6UL * TOK * DMODEL * 4);
  float*  projb    = (float*) alloc((size_t)TOK * PROJ_ * 4);
  float*  simn     = (float*) alloc((size_t)PROJ_ * NL * 4);
  int*    idxb     = (int*)   alloc((size_t)TOK * 4);
  float*  gvalb    = (float*) alloc((size_t)TOK * 4);

  // ---- per-expert buffers: batch as many experts as the workspace allows ----
  const size_t SZ_XLN  = (size_t)TOK * DMODEL * 2;  // bf16
  const size_t SZ_QKV  = (size_t)TOK * 2304 * 4;    // fp32
  const size_t SZ_ATT  = (size_t)TOK * DMODEL * 2;  // bf16
  const size_t SZ_X2   = (size_t)TOK * DMODEL * 4;  // fp32
  const size_t SZ_X2LN = (size_t)TOK * DMODEL * 2;  // bf16
  const size_t SZ_MID  = (size_t)TOK * 3072 * 2;    // bf16
  const size_t PER_E   = SZ_XLN + SZ_QKV + SZ_ATT + SZ_X2 + SZ_X2LN + SZ_MID;
  int batchE = 1;
  {
    const size_t used  = off + 6 * 256;  // + alignment slack
    const size_t avail = (ws_size > used) ? (ws_size - used) : 0;
    long be = (long)(avail / PER_E);
    if (be < 1) be = 1;
    if (be > 6) be = 6;
    batchE = (int)be;   // ws_size is constant across calls -> deterministic
  }
  bf16_t* x_ln_bf  = (bf16_t*)alloc((size_t)batchE * SZ_XLN);
  float*  qkvb     = (float*) alloc((size_t)batchE * SZ_QKV);
  bf16_t* attn_bf  = (bf16_t*)alloc((size_t)batchE * SZ_ATT);
  float*  x2       = (float*) alloc((size_t)batchE * SZ_X2);
  bf16_t* x2_ln_bf = (bf16_t*)alloc((size_t)batchE * SZ_X2LN);
  bf16_t* mid_bf   = (bf16_t*)alloc((size_t)batchE * SZ_MID);

  const long ND    = (long)TOK * DMODEL;   // per-expert activation stride (elems)
  const long NQKV  = (long)TOK * 2304;
  const long NMID  = (long)TOK * 3072;

  // 1) weight transpose+convert (once per launch)
  w_tr_bf16_k<<<dim3(2304 / 32, DMODEL / 32, 6), 256, 0, stream>>>(attn_w,  attn_wt_bf,  DMODEL, 2304);
  w_tr_bf16_k<<<dim3(DMODEL / 32, DMODEL / 32, 6), 256, 0, stream>>>(cproj_w, cproj_wt_bf, DMODEL, DMODEL);
  w_tr_bf16_k<<<dim3(3072 / 32, DMODEL / 32, 6), 256, 0, stream>>>(fc_w,    fc_wt_bf,    DMODEL, 3072);
  w_tr_bf16_k<<<dim3(DMODEL / 32, 3072 / 32, 6), 256, 0, stream>>>(fc2_w,   fc2_wt_bf,   3072, DMODEL);
  cvt_bf16_k<<<1024, 256, 0, stream>>>(gate_w, gate_wt_bf, 5L * PROJ_ * DMODEL);

  // 2) embedding
  embed_k<<<TOK, 256, 0, stream>>>(ids, wte, wpe, h);

  // gE experts (layers j0..j0+gE-1) applied to h, results -> outT[e][TOK,D]
  auto run_blocks = [&](int j0, int gE, float* outT) {
    layernorm_k<<<dim3(TOK, gE), 256, 0, stream>>>(
        h, 0L, ln1_g + (size_t)j0 * DMODEL, ln1_b + (size_t)j0 * DMODEL, (long)DMODEL,
        x_ln_bf, nullptr, ND);
    wmma_gemm_k<<<dim3(2304 / BN, TOK / BM, gE), 256, 0, stream>>>(
        x_ln_bf, ND, attn_wt_bf + (size_t)j0 * DMODEL * 2304, (long)DMODEL * 2304,
        attn_b + (size_t)j0 * 2304, 2304L, nullptr, 0L,
        qkvb, nullptr, NQKV, DMODEL, 2304, 0);
    attn_k<<<dim3(TSEQ / 32, NH, 2 * gE), 256, 0, stream>>>(qkvb, attn_bf);
    wmma_gemm_k<<<dim3(DMODEL / BN, TOK / BM, gE), 256, 0, stream>>>(
        attn_bf, ND, cproj_wt_bf + (size_t)j0 * DMODEL * DMODEL, (long)DMODEL * DMODEL,
        cproj_b + (size_t)j0 * DMODEL, (long)DMODEL, h, 0L,
        x2, nullptr, ND, DMODEL, DMODEL, 0);
    layernorm_k<<<dim3(TOK, gE), 256, 0, stream>>>(
        x2, ND, ln2_g + (size_t)j0 * DMODEL, ln2_b + (size_t)j0 * DMODEL, (long)DMODEL,
        x2_ln_bf, nullptr, ND);
    wmma_gemm_k<<<dim3(3072 / BN, TOK / BM, gE), 256, 0, stream>>>(
        x2_ln_bf, ND, fc_wt_bf + (size_t)j0 * DMODEL * 3072, (long)DMODEL * 3072,
        fc_b + (size_t)j0 * 3072, 3072L, nullptr, 0L,
        nullptr, mid_bf, NMID, DMODEL, 3072, 1 /*gelu*/);
    wmma_gemm_k<<<dim3(DMODEL / BN, TOK / BM, gE), 256, 0, stream>>>(
        mid_bf, NMID, fc2_wt_bf + (size_t)j0 * 3072 * DMODEL, (long)3072 * DMODEL,
        fc2_b + (size_t)j0 * DMODEL, (long)DMODEL, x2, ND,
        outT, nullptr, ND, 3072, DMODEL, 0);
  };

  // 3) MoLEx layers 0..4
  for (int ind = 0; ind < NL - 1; ++ind) {
    const int E = NL - ind;
    cvt_bf16_k<<<512, 256, 0, stream>>>(h, h_bf, ND);
    gate_sim_norm_k<<<1, 256, 0, stream>>>(gate_sim + (size_t)ind * PROJ_ * NL, simn, E);
    wmma_gemm_k<<<dim3(PROJ_ / BN, TOK / BM, 1), 256, 0, stream>>>(
        h_bf, 0L, gate_wt_bf + (size_t)ind * PROJ_ * DMODEL, 0L,
        gate_b + (size_t)ind * PROJ_, 0L, nullptr, 0L,
        projb, nullptr, 0L, DMODEL, PROJ_, 0);
    gate_finalize_k<<<TOK / 8, 256, 0, stream>>>(projb, simn, gate_temp, ind, E,
                                                 idxb, gvalb);
    for (int g0 = 0; g0 < E; g0 += batchE) {
      const int gE = (E - g0 < batchE) ? (E - g0) : batchE;
      run_blocks(ind + g0, gE, outs + (size_t)g0 * TOK * DMODEL);
    }
    combine_k<<<1024, 256, 0, stream>>>(outs, idxb, gvalb, w_main, w_other, ind, h);
  }

  // 4) last block (writes into h; h is no longer read at the fc2 stage)
  run_blocks(NL - 1, 1, h);

  // 5) final layernorm -> fp32 output
  layernorm_k<<<dim3(TOK, 1), 256, 0, stream>>>(h, 0L, lnf_g, lnf_b, 0L,
                                                nullptr, (float*)d_out, 0L);
}
